// SelfAttentionHead_22041772163201
// MI455X (gfx1250) — compile-verified, matrix operand order checked by reading, not run on hardware
//
#include <hip/hip_runtime.h>

typedef __attribute__((ext_vector_type(16))) _Float16 v16h;
typedef __attribute__((ext_vector_type(8)))  _Float16 v8h;
typedef __attribute__((ext_vector_type(4)))  _Float16 v4h;
typedef __attribute__((ext_vector_type(8)))  float    v8f;

#define B_ 8
#define T_ 2048
#define D_ 1024
#define H_ 128

// K index for element e (0..15) of a 16-bit A/B fragment, per CDNA5 ISA 7.12.2:
// lanes 0-15: K = 0..7 (e<8), 16..23 (e>=8); lanes 16-31: +8.
__device__ __forceinline__ int kmap(int e, int half) {
  return e + (half << 3) + ((e >= 8) ? 8 : 0);
}

__device__ __forceinline__ v16h combine(v8h lo, v8h hi) {
  v16h a;
#pragma unroll
  for (int e = 0; e < 8; ++e) { a[e] = lo[e]; a[e + 8] = hi[e]; }
  return a;
}

// A-style fragment (16x32 f16) from row-major f16: lane holds row (lane&15);
// the layout's two contiguous 8-element K runs -> two b128 loads.
__device__ __forceinline__ v16h load_a_f16(const _Float16* __restrict__ src, int ld,
                                           int row0, int k0) {
  int lane = threadIdx.x & 31;
  int half = lane >> 4, m = lane & 15;
  const _Float16* base = src + (size_t)(row0 + m) * ld + k0;
  return combine(*(const v8h*)(base + half * 8),
                 *(const v8h*)(base + 16 + half * 8));
}

#define WMMA_F16(A, Bf, C) \
  __builtin_amdgcn_wmma_f32_16x16x32_f16(false, (A), false, (Bf), (short)0, (C), false, false)

// ---------------- Kernel 0: pack one f32 weight matrix [D_, H_] into B-fragment order ----
// out layout: [kc(0..31)][ntile(0..7)][lane(0..31)][e(0..15)] f16 -> fragment = 1 vector load.
__global__ void __launch_bounds__(256)
pack_w(const float* __restrict__ W, _Float16* __restrict__ out) {
  int i = blockIdx.x * 256 + threadIdx.x;     // 0 .. 131071
  int e    = i & 15;
  int lane = (i >> 4) & 31;
  int nt   = (i >> 9) & 7;
  int kc   = i >> 12;
  int half = lane >> 4, n = lane & 15;
  int krow = kc * 32 + kmap(e, half);
  out[i] = (_Float16)W[(size_t)krow * H_ + nt * 16 + n];
}

// ---------------- Kernel 1: fused q/k/v projection GEMM ----------------
// grid.x = (B*T)/16 row tiles; 8 waves per block each own one 16-col tile of H.
// x tile staged in LDS (contiguous 16x1024 block), weights read as packed fragments.
// v is written directly in attention B-fragment layout.
__global__ void __launch_bounds__(256)
proj_qkv(const float* __restrict__ x,
         const _Float16* __restrict__ wqf, const _Float16* __restrict__ wkf,
         const _Float16* __restrict__ wvf,
         _Float16* __restrict__ q, _Float16* __restrict__ k,
         _Float16* __restrict__ vfrag) {
  __shared__ _Float16 xtile[16 * D_];         // 32 KB
  int row0 = blockIdx.x * 16;                 // flat over B*T

  // cooperative contiguous copy + f32->f16 convert (tile rows are consecutive in x)
  {
    const float4* src = (const float4*)(x + (size_t)row0 * D_);
    v4h* dst = (v4h*)xtile;
    for (int i = threadIdx.x; i < (16 * D_) / 4; i += 256) {
      float4 f = src[i];
      v4h h;
      h[0] = (_Float16)f.x; h[1] = (_Float16)f.y;
      h[2] = (_Float16)f.z; h[3] = (_Float16)f.w;
      dst[i] = h;
    }
  }
  __syncthreads();

  int wave = threadIdx.x >> 5;                // 0..7 -> head-dim tile
  int lane = threadIdx.x & 31;
  int half = lane >> 4, n = lane & 15;
  int col0 = wave * 16;

  v8f accq = {}, acck = {}, accv = {};
  for (int kc = 0; kc < D_ / 32; ++kc) {
    const _Float16* bx = xtile + n * D_ + kc * 32;      // row = lane&15
    v16h xa = combine(*(const v8h*)(bx + half * 8),
                      *(const v8h*)(bx + 16 + half * 8));
    size_t foff = ((size_t)kc * 8 + wave) * 512 + (size_t)lane * 16;
    v16h bq = *(const v16h*)(wqf + foff);
    v16h bk = *(const v16h*)(wkf + foff);
    v16h bv = *(const v16h*)(wvf + foff);
    accq = WMMA_F16(xa, bq, accq);
    acck = WMMA_F16(xa, bk, acck);
    accv = WMMA_F16(xa, bv, accv);
  }

#pragma unroll
  for (int j = 0; j < 8; ++j) {
    int row = row0 + j + half * 8;
    size_t idx = (size_t)row * H_ + col0 + n;
    q[idx] = (_Float16)accq[j];
    k[idx] = (_Float16)acck[j];
    // v -> B-fragment layout: [sc = row/32][hc = wave][lane'][e']
    int ko = row & 31;
    int hp = (ko >> 3) & 1;
    int ep = (ko & 7) + ((ko >= 16) ? 8 : 0);
    int lp = n + (hp << 4);
    vfrag[((size_t)(row >> 5) * 8 + wave) * 512 + lp * 16 + ep] = (_Float16)accv[j];
  }
}

// ---------------- Kernel 2: flash attention ----------------
// Block = 128 consecutive query rows (8 waves x 16 rows). k/v tiles for each
// 32-key chunk are staged once per block into double-buffered LDS and shared
// by all 8 waves (8x cut in L2 traffic vs per-wave streaming).
__global__ void __launch_bounds__(256)
attn_fwd(const _Float16* __restrict__ q, const _Float16* __restrict__ k,
         const _Float16* __restrict__ vfrag, float* __restrict__ out) {
  __shared__ _Float16 ktile[2][32 * H_];   // row-major [key][head], 8 KB per buffer
  __shared__ _Float16 vtile[2][8 * 512];   // fragment order, 8 KB per buffer
  __shared__ _Float16 pbuf[8][16 * 32];    // per-wave P bounce (C-layout -> A-layout)

  int wave = threadIdx.x >> 5;
  int lane = threadIdx.x & 31;
  int half = lane >> 4, n = lane & 15;

  int b   = blockIdx.x >> 4;               // T/128 = 16 blocks per batch
  int t0b = (blockIdx.x & 15) << 7;        // block's first query row
  int t0  = t0b + wave * 16;               // this wave's query tile

  const _Float16* qb  = q + (size_t)b * T_ * H_;
  const _Float16* kb  = k + (size_t)b * T_ * H_;
  const _Float16* vfb = vfrag + (size_t)((b * T_) >> 5) * 4096;

  // q tile 16x128 as 4 A-fragments
  v16h aq[4];
#pragma unroll
  for (int dc = 0; dc < 4; ++dc) aq[dc] = load_a_f16(qb, H_, t0, dc * 32);

  v8f o[8];
#pragma unroll
  for (int hc = 0; hc < 8; ++hc) o[hc] = (v8f){};
  float mrow[8], lrow[8];
#pragma unroll
  for (int j = 0; j < 8; ++j) { mrow[j] = -__builtin_inff(); lrow[j] = 0.0f; }

  int tmax = t0 + 15;
  int smax = t0b + 127;                    // last key any wave in block needs
  for (int s0 = 0; s0 <= smax; s0 += 32) {
    int buf = (s0 >> 5) & 1;

    // ---- cooperative stage: both tiles are contiguous 8 KB runs ----
    {
      int i = threadIdx.x;                 // 256 threads x 16 halves = 4096 halves
      *(v16h*)&ktile[buf][i * 16] = *(const v16h*)(kb + (size_t)s0 * H_ + i * 16);
      *(v16h*)&vtile[buf][i * 16] =
          *(const v16h*)(vfb + (size_t)(s0 >> 5) * 4096 + i * 16);
    }
    __syncthreads();                       // one barrier per chunk (double-buffered)

    if (s0 <= tmax) {                      // wave-uniform causal skip
      // ---- scores: S[16 x 32] = q (16x128) @ k^T (128x32) ----
      v8f slo = {}, shi = {};
#pragma unroll
      for (int dc = 0; dc < 4; ++dc) {
        const _Float16* k0p = &ktile[buf][(lane & 15) * H_ + dc * 32];
        v16h kf0 = combine(*(const v8h*)(k0p + half * 8),
                           *(const v8h*)(k0p + 16 + half * 8));
        const _Float16* k1p = k0p + 16 * H_;
        v16h kf1 = combine(*(const v8h*)(k1p + half * 8),
                           *(const v8h*)(k1p + 16 + half * 8));
        slo = WMMA_F16(aq[dc], kf0, slo);
        shi = WMMA_F16(aq[dc], kf1, shi);
      }

      // ---- causal mask + online softmax (rows j+8*half, col = lane&15) ----
#pragma unroll
      for (int j = 0; j < 8; ++j) {
        int r = t0 + j + half * 8;
        float slv = ((s0 + n)      <= r) ? slo[j] : -__builtin_inff();
        float shv = ((s0 + 16 + n) <= r) ? shi[j] : -__builtin_inff();
        float mx = fmaxf(slv, shv);
        mx = fmaxf(mx, __shfl_xor(mx, 1, 32));
        mx = fmaxf(mx, __shfl_xor(mx, 2, 32));
        mx = fmaxf(mx, __shfl_xor(mx, 4, 32));
        mx = fmaxf(mx, __shfl_xor(mx, 8, 32));
        float mnew  = fmaxf(mrow[j], mx);
        float scale = __expf(mrow[j] - mnew);
        float pl = __expf(slv - mnew);
        float ph = __expf(shv - mnew);
        float rs = pl + ph;
        rs += __shfl_xor(rs, 1, 32);
        rs += __shfl_xor(rs, 2, 32);
        rs += __shfl_xor(rs, 4, 32);
        rs += __shfl_xor(rs, 8, 32);
        lrow[j] = lrow[j] * scale + rs;
        mrow[j] = mnew;
#pragma unroll
        for (int hc = 0; hc < 8; ++hc) o[hc][j] *= scale;
        pbuf[wave][(j + half * 8) * 32 + n]      = (_Float16)pl;
        pbuf[wave][(j + half * 8) * 32 + 16 + n] = (_Float16)ph;
      }

      // stores -> loads within the same wave's LDS slab
      asm volatile("s_wait_dscnt 0" ::: "memory");

      const _Float16* pr = &pbuf[wave][(lane & 15) * 32];
      v16h pa = combine(*(const v8h*)(pr + half * 8),
                        *(const v8h*)(pr + 16 + half * 8));

      // ---- o (16x128) += P (16x32) @ V (32x128), V fragments from LDS ----
#pragma unroll
      for (int hc = 0; hc < 8; ++hc) {
        v16h vf = *(const v16h*)&vtile[buf][hc * 512 + lane * 16];
        o[hc] = WMMA_F16(pa, vf, o[hc]);
      }
    }
  }

  // ---- normalize and store fp32 output ----
#pragma unroll
  for (int j = 0; j < 8; ++j) {
    float inv = 1.0f / lrow[j];
    int r = t0 + j + half * 8;
    size_t base = ((size_t)b * T_ + r) * H_;
#pragma unroll
    for (int hc = 0; hc < 8; ++hc) out[base + hc * 16 + n] = o[hc][j] * inv;
  }
}

extern "C" void kernel_launch(void* const* d_in, const int* in_sizes, int n_in,
                              void* d_out, int out_size, void* d_ws, size_t ws_size,
                              hipStream_t stream) {
  const float* x  = (const float*)d_in[0];
  const float* Wq = (const float*)d_in[1];
  const float* Wk = (const float*)d_in[2];
  const float* Wv = (const float*)d_in[3];
  float* out = (float*)d_out;

  size_t nqkv = (size_t)B_ * T_ * H_;          // 2M elements each
  size_t nwf  = (size_t)D_ * H_;               // 131072 elements per packed weight
  _Float16* qws   = (_Float16*)d_ws;
  _Float16* kws   = qws + nqkv;
  _Float16* vfrag = kws + nqkv;
  _Float16* wqf   = vfrag + nqkv;
  _Float16* wkf   = wqf + nwf;
  _Float16* wvf   = wkf + nwf;

  pack_w<<<(int)(nwf / 256), 256, 0, stream>>>(Wq, wqf);
  pack_w<<<(int)(nwf / 256), 256, 0, stream>>>(Wk, wkf);
  pack_w<<<(int)(nwf / 256), 256, 0, stream>>>(Wv, wvf);
  proj_qkv<<<(B_ * T_) / 16, 256, 0, stream>>>(x, wqf, wkf, wvf, qws, kws, vfrag);
  attn_fwd<<<(B_ * T_) / 128, 256, 0, stream>>>(qws, kws, vfrag, out);
}
